// ChannelFeatureSNN_70454643523773
// MI455X (gfx1250) — matrix operand order repaired and checked by reading, not compiled, not run on hardware
//
#include <hip/hip_runtime.h>

// ---------------------------------------------------------------------------
// Spiking CNN forward on MI455X (gfx1250, wave32, WMMA + TDM).
//   conv2/conv3/conv4/fc1/fc2 -> v_wmma_f32_16x16x32_f16 implicit GEMMs.
//   LIF1 spike trains precomputed bit-packed (cur1 is time-invariant, so the
//   16-step m1 recurrence runs entirely in registers; m1 never touches HBM).
//   LIF fused into all GEMM epilogues. Spike slabs for conv3/conv4/fc staged
//   by the Tensor Data Mover (double-buffered on TENSORcnt) with TDM hardware
//   padding reproducing the bank-skewed 80B LDS row pitch.
// ---------------------------------------------------------------------------

typedef __attribute__((ext_vector_type(16))) _Float16 v16h;
typedef __attribute__((ext_vector_type(8)))  float    v8f;
typedef __attribute__((ext_vector_type(4)))  unsigned int v4u;
typedef __attribute__((ext_vector_type(8)))  int      v8i;
typedef __attribute__((ext_vector_type(4)))  int      v4i;

union V16H { uint4 u[2]; v16h v; };
union H8   { _Float16 h[8]; uint4 u; };
union M8   { unsigned short s[8]; uint4 u; };

#define BETA 0.9f
#define THR  1.0f

#if defined(__has_builtin)
#  if __has_builtin(__builtin_amdgcn_tensor_load_to_lds)
#    define HAVE_TDM 1
#  endif
#endif
#ifndef HAVE_TDM
#  define HAVE_TDM 0
#endif

#if HAVE_TDM
#warning "CDNA5 probe: tensor_load_to_lds builtin AVAILABLE - TDM path enabled"
#else
#warning "CDNA5 probe: tensor_load_to_lds builtin MISSING - VGPR copy fallback"
#endif

// A-fragment K index for 16-bit WMMA (16x32 A): contiguous order, K = 16g + e
// (matches the B fragments read as 16 contiguous halves at column 16g)
__device__ __forceinline__ int kA(int e, int g) {
    return (g << 4) + e;
}

__device__ __forceinline__ unsigned int lds_u32(const void* p) {
    // flat address of an LDS object: addr[31:0] is the LDS offset (ISA 10.2)
    return (unsigned int)(unsigned long long)(uintptr_t)p;
}

#if HAVE_TDM
// One 2D tile: 128 rows x 32 halves, row stride in tensor = stride_halves.
// TDM pad: after every 64B row insert 16B -> LDS row pitch 80B (= CSTR*2).
__device__ __forceinline__ void tdm_load_tile(unsigned int lds_addr, const void* gptr,
                                              unsigned int stride_halves) {
    unsigned long long ga = (unsigned long long)(uintptr_t)gptr;
    v4u g0;
    g0.x = 1u;                                            // count=1, no gather
    g0.y = lds_addr;                                      // lds_addr (bytes)
    g0.z = (unsigned int)(ga & 0xFFFFFFFFu);              // global_addr[31:0]
    g0.w = (unsigned int)((ga >> 32) & 0x01FFFFFFu) | (2u << 30);  // [56:32] | type=2
    v8i g1;
    g1[0] = (int)((1u << 16) |      // data_size = 2 bytes
                  (1u << 20) |      // pad_enable
                  (3u << 22) |      // pad_interval: every 16 DWORDs (64B)
                  (3u << 25));      // pad_amount: 4 DWORDs (16B)
    g1[1] = (int)(32u << 16);       // tensor_dim0 = 32 (lo16 @ bits 63:48)
    g1[2] = (int)(128u << 16);      // tensor_dim0 hi=0 | tensor_dim1 lo16 = 128
    g1[3] = (int)(32u << 16);       // tensor_dim1 hi=0 | tile_dim0 = 32
    g1[4] = (int)128u;              // tile_dim1 = 128, tile_dim2 = 0
    g1[5] = (int)stride_halves;     // tensor_dim0_stride lo32 (data_size units)
    g1[6] = 0;                      // stride hi16 | tensor_dim1_stride lo16
    g1[7] = 0;
    v4i gz = {0, 0, 0, 0};
#if __clang_major__ >= 23
    v8i gz8 = {0, 0, 0, 0, 0, 0, 0, 0};
    __builtin_amdgcn_tensor_load_to_lds(g0, g1, gz, gz, gz8, 0);
#else
    __builtin_amdgcn_tensor_load_to_lds(g0, g1, gz, gz, 0);
#endif
}
#endif

// ---------------------------------------------------------------------------
// Weight pre-pack into exact WMMA A-fragment lane/element order.
// mode 0: conv W[co][k][dk]  mode 1: fc W[co][k]  mode 2: fc1 + flatten perm.
// ---------------------------------------------------------------------------
__global__ void pack_kernel(const float* __restrict__ W, _Float16* __restrict__ out,
                            int total, int nkc, int KT, int COUT, int KIN, int mode) {
    int idx = blockIdx.x * 256 + threadIdx.x;
    if (idx >= total) return;
    int e    = idx & 15;
    int lane = (idx >> 4) & 31;
    int rest = idx >> 9;
    int dk = rest % KT;  rest /= KT;
    int kc = rest % nkc; int grp = rest / nkc;
    int m = lane & 15, g = lane >> 4;
    int co = grp * 16 + m;
    int k  = kc * 32 + kA(e, g);
    float v = 0.0f;
    if (co < COUT && k < KIN) {
        if (mode == 0)      v = W[((size_t)co * KIN + k) * KT + dk];
        else if (mode == 1) v = W[(size_t)co * KIN + k];
        else                v = W[(size_t)co * 256 + ((k & 1) * 128 + (k >> 1))];
    }
    out[idx] = (_Float16)v;
}

// ---------------------------------------------------------------------------
// conv1 + 16-step LIF1 entirely in registers -> bit-packed spike trains.
// cur1 is time-invariant, so each element's m1 trajectory is independent.
// s1t[(b*128+l)*256 + co] bit t = spike at timestep t. Exact f32 math.
// ---------------------------------------------------------------------------
__global__ __launch_bounds__(256)
void spiketrain_kernel(const float* __restrict__ x, const float* __restrict__ W1,
                       const float* __restrict__ b1, unsigned short* __restrict__ s1t)
{
    __shared__ float xs[256];
    const int b = blockIdx.x, tid = threadIdx.x;
    xs[tid] = x[(size_t)b * 256 + tid];
    __syncthreads();
    float w[10];
    #pragma unroll
    for (int i = 0; i < 10; ++i) w[i] = W1[tid * 10 + i];
    const float bb = b1[tid];
    for (int l = 0; l < 128; ++l) {
        float cur = bb;
        #pragma unroll
        for (int ci = 0; ci < 2; ++ci)
            #pragma unroll
            for (int kk = 0; kk < 5; ++kk) {
                int li = l + kk - 2;
                if (li >= 0 && li < 128) cur += xs[ci * 128 + li] * w[ci * 5 + kk];
            }
        float m = 0.0f;
        unsigned int bits = 0;
        #pragma unroll
        for (int t = 0; t < 16; ++t) {
            float rs = (m > THR) ? 1.0f : 0.0f;          // reset from previous mem
            m = BETA * m + cur - rs * THR;
            bits |= (((m - THR) > 0.0f) ? 1u : 0u) << t;
        }
        s1t[(size_t)(b * 128 + l) * 256 + tid] = (unsigned short)bits;
    }
}

// ---------------------------------------------------------------------------
// Fused conv(K=3,pad=1) + LIF, implicit GEMM per batch.
//   MODE 0: fp16 spike input (TDM staging)            (conv3)
//   MODE 2: like 0, flat (B,256) output               (conv4)
//   MODE 3: bit-extract of s1t spike trains in fill   (conv2)
// ---------------------------------------------------------------------------
template<int CIN, int COUT, int COUTPAD, int BDIM, int MODE>
__global__ __launch_bounds__(BDIM)
void conv_lif_kernel(const _Float16* __restrict__ Wp,
                     const float*    __restrict__ bias,
                     const _Float16* __restrict__ sp_in,      // MODE 0/2
                     const unsigned short* __restrict__ s1t,  // MODE 3
                     int             tstep,                   // MODE 3
                     float*          __restrict__ m_out,
                     _Float16*       __restrict__ s_out)
{
    constexpr int NKC  = CIN / 32;
    constexpr int NCW  = COUTPAD / 16;
    constexpr int CSTR = 40;                  // LDS row pitch (halves) = 80B
    constexpr int SLAB = 130 * CSTR;          // rows l=-1..128
#if HAVE_TDM
    constexpr int NBUF = (MODE == 3) ? 1 : 2;
#else
    constexpr int NBUF = 1;
#endif

    __shared__ _Float16 slab[NBUF * SLAB];
    __shared__ float    sbias[COUTPAD];

    const int b = blockIdx.x, tid = threadIdx.x;
    const int lane = tid & 31, wave = tid >> 5;
    const int g = lane >> 4, n = lane & 15;

    // zero the padded boundary rows of every buffer (stay zero afterwards)
    for (int i = tid; i < 8 * NBUF; i += BDIM) {
        int bu = i >> 3, r = i & 7;
        int row = (r & 1) ? 129 : 0;
        int c8  = (r >> 1) * 8;
        H8 z;
        #pragma unroll
        for (int j = 0; j < 8; ++j) z.h[j] = (_Float16)0.0f;
        *(uint4*)&slab[bu * SLAB + row * CSTR + c8] = z.u;
    }
    for (int i = tid; i < COUTPAD; i += BDIM) sbias[i] = (i < COUT) ? bias[i] : 0.0f;
    __syncthreads();

    v8f acc[8];
    #pragma unroll
    for (int f = 0; f < 8; ++f)
        #pragma unroll
        for (int j = 0; j < 8; ++j) acc[f][j] = 0.0f;

    // GEMM step: all A taps + all 8 B fragments batched before the WMMAs
    auto compute_chunk = [&](const _Float16* sb, int kci) {
        if (wave < NCW) {
            V16H a[3];
            #pragma unroll
            for (int dk = 0; dk < 3; ++dk) {
                const uint4* ap = (const uint4*)&Wp[(((size_t)wave * NKC + kci) * 3 + dk) * 512 + lane * 16];
                a[dk].u[0] = ap[0]; a[dk].u[1] = ap[1];
            }
            #pragma unroll
            for (int dk = 0; dk < 3; ++dk) {
                V16H bm[8];
                #pragma unroll
                for (int nf = 0; nf < 8; ++nf) {
                    const uint4* bp = (const uint4*)&sb[(nf * 16 + n + dk) * CSTR + g * 16];
                    bm[nf].u[0] = bp[0]; bm[nf].u[1] = bp[1];
                }
                #pragma unroll
                for (int nf = 0; nf < 8; ++nf)
                    acc[nf] = __builtin_amdgcn_wmma_f32_16x16x32_f16(
                        false, a[dk].v, false, bm[nf].v, (short)0, acc[nf], false, false);
            }
        }
    };

    if constexpr (MODE == 3) {
        // fill = bit-extract of the precomputed LIF1 spike train (tiny traffic)
        for (int kci = 0; kci < NKC; ++kci) {
            const int kc = kci * 32;
            for (int u = tid; u < 512; u += BDIM) {
                const int l = u >> 2, c8 = (u & 3) * 8;
                const size_t gi = (size_t)(b * 128 + l) * 256 + kc + c8;
                M8 ms;
                ms.u = *(const uint4*)&s1t[gi];
                H8 sp;
                #pragma unroll
                for (int j = 0; j < 8; ++j)
                    sp.h[j] = (_Float16)(int)((ms.s[j] >> tstep) & 1u);
                *(uint4*)&slab[(l + 1) * CSTR + c8] = sp.u;
            }
            __syncthreads();
            compute_chunk(slab, kci);
            __syncthreads();
        }
    } else {
#if HAVE_TDM
        // TDM-staged, double-buffered on TENSORcnt
        if (wave == 0)
            tdm_load_tile(lds_u32(&slab[CSTR]), sp_in + (size_t)b * 128 * CIN, CIN);
        for (int kci = 0; kci < NKC; ++kci) {
            if (wave == 0) {
                if (kci + 1 < NKC) {
                    tdm_load_tile(lds_u32(&slab[((kci + 1) & 1) * SLAB + CSTR]),
                                  sp_in + (size_t)b * 128 * CIN + (kci + 1) * 32, CIN);
                    __builtin_amdgcn_s_wait_tensorcnt(1);
                } else {
                    __builtin_amdgcn_s_wait_tensorcnt(0);
                }
            }
            __syncthreads();
            compute_chunk(&slab[(kci & 1) * SLAB], kci);
            __syncthreads();
        }
#else
        for (int kci = 0; kci < NKC; ++kci) {
            const int kc = kci * 32;
            for (int u = tid; u < 512; u += BDIM) {
                const int l = u >> 2, c8 = (u & 3) * 8;
                const size_t gi = (size_t)(b * 128 + l) * CIN + kc + c8;
                *(uint4*)&slab[(l + 1) * CSTR + c8] = *(const uint4*)&sp_in[gi];
            }
            __syncthreads();
            compute_chunk(slab, kci);
            __syncthreads();
        }
#endif
    }

    // ---- fused LIF epilogue ----
    if (wave < NCW) {
        const int cobase = wave * 16 + g * 8;       // C frag: M = r + 8g
        #pragma unroll
        for (int nf = 0; nf < 8; ++nf) {
            const int l = nf * 16 + n;
            if constexpr (MODE == 2) {              // conv4: COUT=2, flat (l*2+c)
                #pragma unroll
                for (int r = 0; r < 8; ++r) {
                    const int co = cobase + r;
                    if (co < COUT) {
                        float cur = acc[nf][r] + sbias[co];
                        const size_t mi = (size_t)(b * 128 + l) * COUT + co;
                        float mo = m_out[mi];
                        float rs = (mo > THR) ? 1.0f : 0.0f;
                        float mn = BETA * mo + cur - rs * THR;
                        m_out[mi] = mn;
                        s_out[(size_t)b * 256 + l * 2 + co] =
                            (_Float16)(((mn - THR) > 0.0f) ? 1.0f : 0.0f);
                    }
                }
            } else {
                const size_t mi = (size_t)(b * 128 + l) * COUT + cobase;
                float mm[8];
                *(float4*)&mm[0] = *(const float4*)&m_out[mi];
                *(float4*)&mm[4] = *(const float4*)&m_out[mi + 4];
                H8 sp;
                #pragma unroll
                for (int r = 0; r < 8; ++r) {
                    float cur = acc[nf][r] + sbias[cobase + r];
                    float mo = mm[r];
                    float rs = (mo > THR) ? 1.0f : 0.0f;
                    float mn = BETA * mo + cur - rs * THR;
                    mm[r] = mn;
                    sp.h[r] = (_Float16)(((mn - THR) > 0.0f) ? 1.0f : 0.0f);
                }
                *(float4*)&m_out[mi]     = *(float4*)&mm[0];
                *(float4*)&m_out[mi + 4] = *(float4*)&mm[4];
                *(uint4*)&s_out[mi]      = sp.u;
            }
        }
    }
}

// ---------------------------------------------------------------------------
// FC GEMM: M = out features (padded to 16), N = 128 batches/block, K = KDIM.
// ---------------------------------------------------------------------------
template<int KDIM, int MOUT, int MPAD, int BDIM, bool DO_LIF>
__global__ __launch_bounds__(BDIM)
void fc_kernel(const _Float16* __restrict__ Wp, const float* __restrict__ bias,
               const _Float16* __restrict__ in, float* __restrict__ mstate,
               _Float16* __restrict__ s_out, float* __restrict__ out_acc)
{
    constexpr int NKC  = KDIM / 32;
    constexpr int NCW  = MPAD / 16;
    constexpr int CSTR = 40;
    constexpr int SLAB = 128 * CSTR;
#if HAVE_TDM
    constexpr int NBUF = 2;
#else
    constexpr int NBUF = 1;
#endif
    __shared__ _Float16 slab[NBUF * SLAB];
    __shared__ float sbias[MPAD];

    const int bbase = blockIdx.x * 128, tid = threadIdx.x;
    const int lane = tid & 31, wave = tid >> 5;
    const int g = lane >> 4, n = lane & 15;

    for (int i = tid; i < MPAD; i += BDIM) sbias[i] = (i < MOUT) ? bias[i] : 0.0f;
    __syncthreads();

    v8f acc[8];
    #pragma unroll
    for (int f = 0; f < 8; ++f)
        #pragma unroll
        for (int j = 0; j < 8; ++j) acc[f][j] = 0.0f;

    auto compute_chunk = [&](const _Float16* sb, int kci) {
        if (wave < NCW) {
            V16H a;
            const uint4* ap = (const uint4*)&Wp[((size_t)wave * NKC + kci) * 512 + lane * 16];
            a.u[0] = ap[0]; a.u[1] = ap[1];
            V16H bm[8];
            #pragma unroll
            for (int nf = 0; nf < 8; ++nf) {
                const uint4* bp = (const uint4*)&sb[(nf * 16 + n) * CSTR + g * 16];
                bm[nf].u[0] = bp[0]; bm[nf].u[1] = bp[1];
            }
            #pragma unroll
            for (int nf = 0; nf < 8; ++nf)
                acc[nf] = __builtin_amdgcn_wmma_f32_16x16x32_f16(
                    false, a.v, false, bm[nf].v, (short)0, acc[nf], false, false);
        }
    };

#if HAVE_TDM
    if (wave == 0)
        tdm_load_tile(lds_u32(&slab[0]), in + (size_t)bbase * KDIM, KDIM);
    for (int kci = 0; kci < NKC; ++kci) {
        if (wave == 0) {
            if (kci + 1 < NKC) {
                tdm_load_tile(lds_u32(&slab[((kci + 1) & 1) * SLAB]),
                              in + (size_t)bbase * KDIM + (kci + 1) * 32, KDIM);
                __builtin_amdgcn_s_wait_tensorcnt(1);
            } else {
                __builtin_amdgcn_s_wait_tensorcnt(0);
            }
        }
        __syncthreads();
        compute_chunk(&slab[(kci & 1) * SLAB], kci);
        __syncthreads();
    }
#else
    for (int kci = 0; kci < NKC; ++kci) {
        const int kc = kci * 32;
        for (int u = tid; u < 512; u += BDIM) {
            const int bl = u >> 2, off = (u & 3) * 8;
            *(uint4*)&slab[bl * CSTR + off] =
                *(const uint4*)&in[(size_t)(bbase + bl) * KDIM + kc + off];
        }
        __syncthreads();
        compute_chunk(slab, kci);
        __syncthreads();
    }
#endif

    if (wave < NCW) {
        const int cobase = wave * 16 + g * 8;
        #pragma unroll
        for (int nf = 0; nf < 8; ++nf) {
            const int bidx = bbase + nf * 16 + n;
            #pragma unroll
            for (int r = 0; r < 8; ++r) {
                const int co = cobase + r;
                if constexpr (DO_LIF) {
                    float cur = acc[nf][r] + sbias[co];
                    const size_t mi = (size_t)bidx * 128 + co;   // padded stride
                    float mo = mstate[mi];
                    float rs = (mo > THR) ? 1.0f : 0.0f;
                    float mn = BETA * mo + cur - rs * THR;
                    mstate[mi] = mn;
                    s_out[mi] = (_Float16)((co < MOUT && (mn - THR) > 0.0f) ? 1.0f : 0.0f);
                } else {
                    if (co < MOUT)
                        out_acc[(size_t)bidx * MOUT + co] += acc[nf][r] + sbias[co];
                }
            }
        }
    }
}

__global__ void scale_kernel(float* __restrict__ out, int nelem) {
    int i = blockIdx.x * 256 + threadIdx.x;
    if (i < nelem) out[i] *= (1.0f / 16.0f);
}

// ---------------------------------------------------------------------------
extern "C" void kernel_launch(void* const* d_in, const int* in_sizes, int n_in,
                              void* d_out, int out_size, void* d_ws, size_t ws_size,
                              hipStream_t stream) {
    const float* x   = (const float*)d_in[0];
    const float* W1  = (const float*)d_in[1];
    const float* b1  = (const float*)d_in[2];
    const float* W2  = (const float*)d_in[3];
    const float* b2  = (const float*)d_in[4];
    const float* W3  = (const float*)d_in[5];
    const float* b3  = (const float*)d_in[6];
    const float* W4  = (const float*)d_in[7];
    const float* b4  = (const float*)d_in[8];
    const float* fw1 = (const float*)d_in[9];
    const float* fb1 = (const float*)d_in[10];
    const float* fw2 = (const float*)d_in[11];
    const float* fb2 = (const float*)d_in[12];
    float* out = (float*)d_out;

    char* base = (char*)d_ws;
    size_t off = 0;
    auto take = [&](size_t bytes) -> char* {
        char* p = base + off;
        off += (bytes + 255) & ~(size_t)255;
        return p;
    };
    const size_t SZ_S1T = 2048UL * 128 * 256 * 2;   // bit-packed LIF1 spike trains
    const size_t SZ_M2  = 2048UL * 128 * 128 * 4;
    const size_t SZ_M3  = 2048UL * 128 * 64 * 4;
    const size_t SZ_M4  = 2048UL * 128 * 2 * 4;
    const size_t SZ_M5  = 2048UL * 128 * 4;
    const size_t SZ_S2  = 2048UL * 128 * 128 * 2;
    const size_t SZ_S3  = 2048UL * 128 * 64 * 2;
    const size_t SZ_S4  = 2048UL * 256 * 2;
    const size_t SZ_S5  = 2048UL * 128 * 2;

    unsigned short* s1t = (unsigned short*)take(SZ_S1T);
    float*     m2  = (float*)take(SZ_M2);
    float*     m3  = (float*)take(SZ_M3);
    float*     m4  = (float*)take(SZ_M4);
    float*     m5  = (float*)take(SZ_M5);
    _Float16*  s2  = (_Float16*)take(SZ_S2);
    _Float16*  s3  = (_Float16*)take(SZ_S3);
    _Float16*  s4  = (_Float16*)take(SZ_S4);
    _Float16*  s5  = (_Float16*)take(SZ_S5);
    _Float16*  W2p = (_Float16*)take(98304UL * 2);
    _Float16*  W3p = (_Float16*)take(24576UL * 2);
    _Float16*  W4p = (_Float16*)take(3072UL * 2);
    _Float16*  f1p = (_Float16*)take(28672UL * 2);
    _Float16*  f2p = (_Float16*)take(6144UL * 2);

    hipMemsetAsync(m2, 0, SZ_M2, stream);
    hipMemsetAsync(m3, 0, SZ_M3, stream);
    hipMemsetAsync(m4, 0, SZ_M4, stream);
    hipMemsetAsync(m5, 0, SZ_M5, stream);
    hipMemsetAsync(s5, 0, SZ_S5, stream);   // K-pad region must stay zero
    hipMemsetAsync(out, 0, (size_t)out_size * 4, stream);

    pack_kernel<<<(98304 + 255) / 256, 256, 0, stream>>>(W2,  W2p, 98304, 8, 3, 128, 256, 0);
    pack_kernel<<<(24576 + 255) / 256, 256, 0, stream>>>(W3,  W3p, 24576, 4, 3, 64, 128, 0);
    pack_kernel<<<(3072  + 255) / 256, 256, 0, stream>>>(W4,  W4p, 3072,  2, 3, 2,  64,  0);
    pack_kernel<<<(28672 + 255) / 256, 256, 0, stream>>>(fw1, f1p, 28672, 8, 1, 100, 256, 2);
    pack_kernel<<<(6144  + 255) / 256, 256, 0, stream>>>(fw2, f2p, 6144,  4, 1, 40, 100, 1);

    // conv1 + all 16 LIF1 steps in registers -> bit-packed spike trains
    spiketrain_kernel<<<2048, 256, 0, stream>>>(x, W1, b1, s1t);

    for (int t = 0; t < 16; ++t) {
        conv_lif_kernel<256, 128, 128, 256, 3><<<2048, 256, 0, stream>>>(
            W2p, b2, nullptr, s1t, t, m2, s2);
        conv_lif_kernel<128, 64, 64, 128, 0><<<2048, 128, 0, stream>>>(
            W3p, b3, s2, nullptr, 0, m3, s3);
        conv_lif_kernel<64, 2, 16, 64, 2><<<2048, 64, 0, stream>>>(
            W4p, b4, s3, nullptr, 0, m4, s4);
        fc_kernel<256, 100, 112, 224, true><<<16, 224, 0, stream>>>(
            f1p, fb1, s4, m5, s5, nullptr);
        fc_kernel<128, 40, 48, 96, false><<<16, 96, 0, stream>>>(
            f2p, fb2, s5, nullptr, nullptr, out);
    }
    scale_kernel<<<(out_size + 255) / 256, 256, 0, stream>>>(out, out_size);
}
